// BidirRecurrentModel_64398739636720
// MI455X (gfx1250) — matrix operand order repaired and checked by reading, not compiled, not run on hardware
//
#include <hip/hip_runtime.h>
#include <cstddef>
#include <cstdint>

// Problem constants (from reference): B=32, T=128, I=512, H=512, L=2, O=512
#define T_STEPS 128
#define NB      32
#define IDIM    512
#define HDIM    512
#define H2DIM   1024
#define ODIM    512
#define ROWS    (T_STEPS * NB)   // 4096 time-major rows

typedef _Float16 h8   __attribute__((ext_vector_type(8)));
typedef _Float16 v16h __attribute__((ext_vector_type(16)));
typedef float    v8f  __attribute__((ext_vector_type(8)));

// ---------------------------------------------------------------------------
// CDNA5 async global->LDS copy (ASYNCcnt path, no VGPR round trip).
// GVS form: mem = SADDR(64) + VADDR(u32 byte offset); dsaddr = VDST VGPR.
// LDS byte offset = low 32 bits of the generic pointer to __shared__ memory.
// ---------------------------------------------------------------------------
__device__ __forceinline__ void async_copy16(void* lds_ptr, const void* gbase,
                                             uint32_t gbyte_off) {
  uint32_t lds_off = (uint32_t)(uintptr_t)lds_ptr;
  asm volatile("global_load_async_to_lds_b128 %0, %1, %2"
               :
               : "v"(lds_off), "v"(gbyte_off), "s"(gbase)
               : "memory");
}
__device__ __forceinline__ void wait_async0() {
  asm volatile("s_wait_asynccnt 0" ::: "memory");
}

// ---------------------------------------------------------------------------
// WMMA fragment helpers (layouts per CDNA5 ISA 7.12.2, wave32)
// ---------------------------------------------------------------------------
__device__ __forceinline__ v8f wmma16(v16h a, v16h b, v8f c) {
  return __builtin_amdgcn_wmma_f32_16x16x32_f16(false, a, false, b, (short)0, c,
                                                false, false);
}

// A fragment (16x32 f16) from row-major [M][ld] storage.
__device__ __forceinline__ v16h frag_a_rm(const _Float16* __restrict__ base,
                                          int ld, int m0, int k0, int lane) {
  const int half = (lane >> 4) & 1;
  const int row  = m0 + (lane & 15);
  const _Float16* p = base + (size_t)row * ld + k0 + 8 * half;
  h8 lo = *(const h8*)(p);
  h8 hi = *(const h8*)(p + 16);
  v16h f;
#pragma unroll
  for (int i = 0; i < 8; ++i) { f[i] = lo[i]; f[i + 8] = hi[i]; }
  return f;
}

// B fragment (32x16 f16) from column-major [N][ld] storage.
__device__ __forceinline__ v16h frag_b_cm(const _Float16* __restrict__ base,
                                          int ld, int n0, int k0, int lane) {
  const int half = (lane >> 4) & 1;
  const int col  = n0 + (lane & 15);
  const _Float16* p = base + (size_t)col * ld + k0 + 8 * half;
  h8 lo = *(const h8*)(p);
  h8 hi = *(const h8*)(p + 16);
  v16h f;
#pragma unroll
  for (int i = 0; i < 8; ++i) { f[i] = lo[i]; f[i + 8] = hi[i]; }
  return f;
}

__device__ __forceinline__ float sigmoidf_dev(float x) {
  return 1.0f / (1.0f + __expf(-x));
}

// ---------------------------------------------------------------------------
// Pack kernels
// ---------------------------------------------------------------------------
__global__ void pack_x(const float* __restrict__ x,
                       _Float16* __restrict__ seqF,
                       _Float16* __restrict__ seqR) {
  int i = blockIdx.x * blockDim.x + threadIdx.x;   // over T*B*I
  if (i >= T_STEPS * NB * IDIM) return;
  int c  = i & (IDIM - 1);
  int bt = i >> 9;               // t*NB + b
  int b  = bt & (NB - 1);
  int t  = bt >> 5;
  seqF[i] = (_Float16)x[((size_t)b * T_STEPS + t) * IDIM + c];
  seqR[i] = (_Float16)x[((size_t)b * T_STEPS + (T_STEPS - 1 - t)) * IDIM + c];
}

// f32 [K][N] row-major -> f16 [N][K] column-major (transpose + convert)
__global__ void pack_wT(const float* __restrict__ src,
                        _Float16* __restrict__ dst, int K, int N) {
  int i = blockIdx.x * blockDim.x + threadIdx.x;
  if (i >= K * N) return;
  int n = i % N, k = i / N;
  dst[(size_t)n * K + k] = (_Float16)src[i];
}

__global__ void init_ctr(unsigned* c) {
  if (threadIdx.x < 8) c[threadIdx.x] = 0u;
}

// ---------------------------------------------------------------------------
// Time-parallel input projection GEMM:
//   C[M][N] f32 = A[M][K] f16 (row-major) * Wt[N][K] f16 (col-major) + bias[N]
// Block tile 64(M) x 128(N), 256 threads = 8 waves (2x4), kstep = 32.
// LDS tiles filled with async global->LDS copies (ASYNCcnt).
// ---------------------------------------------------------------------------
__global__ __launch_bounds__(256, 2) void gemm_xproj(
    const _Float16* __restrict__ A, const _Float16* __restrict__ Wt,
    const float* __restrict__ bias, float* __restrict__ C,
    int M, int N, int K) {
  __shared__ _Float16 sA[64][40];    // +8 pad against bank conflicts
  __shared__ _Float16 sB[128][40];

  const int tid  = threadIdx.x;
  const int lane = tid & 31;
  const int wave = tid >> 5;
  const int wm = wave >> 2;          // 0..1  (M direction)
  const int wn = wave & 3;           // 0..3  (N direction)
  const int mBase = blockIdx.y * 64;
  const int nBase = blockIdx.x * 128;

  const int rowA = tid >> 2, segA = tid & 3;       // A: 8 f16 per thread
  const int colB = tid >> 1, segB = tid & 1;       // B: 16 f16 per thread
  const uint32_t offA0 = (uint32_t)((mBase + rowA) * K + segA * 8) * 2u;
  const uint32_t offB0 = (uint32_t)((nBase + colB) * K + segB * 16) * 2u;

  v8f acc[2][2] = {};

  for (int k0 = 0; k0 < K; k0 += 32) {
    const uint32_t kb = (uint32_t)k0 * 2u;
    async_copy16(&sA[rowA][segA * 8],       A,  offA0 + kb);
    async_copy16(&sB[colB][segB * 16],      Wt, offB0 + kb);
    async_copy16(&sB[colB][segB * 16 + 8],  Wt, offB0 + kb + 16u);
    wait_async0();
    __syncthreads();

    v16h af[2], bf[2];
#pragma unroll
    for (int mi = 0; mi < 2; ++mi)
      af[mi] = frag_a_rm(&sA[0][0], 40, wm * 32 + mi * 16, 0, lane);
#pragma unroll
    for (int nj = 0; nj < 2; ++nj)
      bf[nj] = frag_b_cm(&sB[0][0], 40, wn * 32 + nj * 16, 0, lane);
#pragma unroll
    for (int mi = 0; mi < 2; ++mi)
#pragma unroll
      for (int nj = 0; nj < 2; ++nj)
        acc[mi][nj] = wmma16(af[mi], bf[nj], acc[mi][nj]);
    __syncthreads();
  }

  // Epilogue: C mapping: m = m0 + v + 8*(lane>=16), n = n0 + lane%16
#pragma unroll
  for (int mi = 0; mi < 2; ++mi) {
#pragma unroll
    for (int nj = 0; nj < 2; ++nj) {
      int m0 = mBase + wm * 32 + mi * 16;
      int n  = nBase + wn * 32 + nj * 16 + (lane & 15);
      float bv = bias[n];
      float* cp = C + (size_t)(m0 + 8 * (lane >> 4)) * N + n;
#pragma unroll
      for (int v = 0; v < 8; ++v) cp[(size_t)v * N] = acc[mi][nj][v] + bv;
    }
  }
}

// ---------------------------------------------------------------------------
// Persistent recurrent kernel: one launch per layer, both directions, all T
// steps.  32 workgroups (16 per direction), 192 threads (6 waves) each.
// Each WG owns hidden units [j0, j0+32): weight slice (z|r|g = 96 cols x 512)
// resident in LDS (96 KB, f16, filled via async copies).  Per step: h staged
// in LDS via async copies, 6x2 WMMA tiles, fused GRU elementwise, h exchanged
// through L2 via ping-pong buffer + device-scope atomic barrier.
// ---------------------------------------------------------------------------
#define GRU_NWG_DIR 16
#define GRU_NWG     (2 * GRU_NWG_DIR)
#define GRU_THREADS 192
#define GRU_SMEM_BYTES ((96 * 512 + 32 * 512) * 2 + 96 * 33 * 4)  // 143744

__global__ __launch_bounds__(GRU_THREADS, 1) void gru_step_all(
    const _Float16* __restrict__ WhhT,   // [2][1024][512] (this layer)
    const _Float16* __restrict__ WhrT,   // [2][512][512]
    const float* __restrict__ bhh,       // [2][1024]
    const float* __restrict__ bhr,       // [2][512]
    const float* __restrict__ XH,        // [2][T*B][1024]  (x@Wxh + bxh)
    const float* __restrict__ XR,        // [2][T*B][512]   (x@Wxr + bxr)
    _Float16* __restrict__ Yout,         // [2][T*B][512]
    _Float16* __restrict__ Hbuf,         // [2][2][32*512] ping-pong per dir
    unsigned* __restrict__ ctr, int T) {
  extern __shared__ char smem[];
  _Float16* lW = (_Float16*)smem;                 // [96][512] z|r|g weights
  _Float16* lH = lW + 96 * 512;                   // [32][512] current h
  float*    lS = (float*)(lH + 32 * 512);         // [96][33]  pre-activations

  const int dir   = blockIdx.x >> 4;
  const int slice = blockIdx.x & 15;
  const int j0    = slice * 32;
  const int tid   = threadIdx.x;
  const int lane  = tid & 31;
  const int wave  = tid >> 5;                     // 0..5 -> column tile

  // Resident weights: rows 0-31 = z (Whh[:,j0..]), 32-63 = r (Whh[:,512+j0..]),
  // 64-95 = g (Whr[:,j0..]).  Async global->LDS, 16B chunks.
  {
    const _Float16* s0 = WhhT + (size_t)dir * H2DIM * HDIM + (size_t)j0 * HDIM;
    const _Float16* s1 = WhhT + (size_t)dir * H2DIM * HDIM + (size_t)(512 + j0) * HDIM;
    const _Float16* s2 = WhrT + (size_t)dir * HDIM * HDIM + (size_t)j0 * HDIM;
    for (int i = tid; i < (32 * 512) / 8; i += GRU_THREADS)
      async_copy16(lW + 0 * 32 * 512 + i * 8, s0, (uint32_t)i * 16u);
    for (int i = tid; i < (32 * 512) / 8; i += GRU_THREADS)
      async_copy16(lW + 1 * 32 * 512 + i * 8, s1, (uint32_t)i * 16u);
    for (int i = tid; i < (32 * 512) / 8; i += GRU_THREADS)
      async_copy16(lW + 2 * 32 * 512 + i * 8, s2, (uint32_t)i * 16u);
  }
  // h0 = 0
  for (int i = tid; i < 32 * 512; i += GRU_THREADS) lH[i] = (_Float16)0.0f;
  wait_async0();
  __syncthreads();

  const float*    XHd   = XH + (size_t)dir * T * NB * H2DIM;
  const float*    XRd   = XR + (size_t)dir * T * NB * HDIM;
  _Float16*       Yd    = Yout + (size_t)dir * T * NB * HDIM;
  _Float16*       Hping = Hbuf + (size_t)dir * 2 * NB * HDIM;
  const float*    bz    = bhh + (size_t)dir * H2DIM;
  const float*    bg    = bhr + (size_t)dir * HDIM;

  for (int t = 0; t < T; ++t) {
    if (t > 0) {  // refresh full h (32 KB) from last step's buffer, async
      const _Float16* hsrc = Hping + (size_t)((t - 1) & 1) * NB * HDIM;
      for (int i = tid; i < (NB * HDIM) / 8; i += GRU_THREADS)
        async_copy16(lH + i * 8, hsrc, (uint32_t)i * 16u);
      wait_async0();
      __syncthreads();
    }

    // h @ W for this wave's 16-column tile, both 16-row halves of the batch
    v8f acc0 = {}, acc1 = {};
#pragma unroll
    for (int ks = 0; ks < HDIM; ks += 32) {
      v16h b  = frag_b_cm(lW, 512, wave * 16, ks, lane);
      v16h a0 = frag_a_rm(lH, 512, 0, ks, lane);
      v16h a1 = frag_a_rm(lH, 512, 16, ks, lane);
      acc0 = wmma16(a0, b, acc0);
      acc1 = wmma16(a1, b, acc1);
    }

    // Stage pre-activations: lS[c][m], c = wave*16 + lane%16
    {
      int c  = wave * 16 + (lane & 15);
      int mb = 8 * (lane >> 4);
#pragma unroll
      for (int v = 0; v < 8; ++v) {
        lS[c * 33 + mb + v]      = acc0[v];
        lS[c * 33 + 16 + mb + v] = acc1[v];
      }
    }
    __syncthreads();

    // Fused GRU elementwise for owned units (32 batch x 32 hidden)
    for (int i = tid; i < NB * 32; i += GRU_THREADS) {
      int jj = i & 31;            // fastest -> coalesced stores
      int b  = i >> 5;
      int j  = j0 + jj;
      size_t rb = (size_t)t * NB + b;
      float zp = lS[jj * 33 + b]        + XHd[rb * H2DIM + j]       + bz[j];
      float rp = lS[(32 + jj) * 33 + b] + XHd[rb * H2DIM + 512 + j] + bz[512 + j];
      float gp = lS[(64 + jj) * 33 + b] + bg[j];
      float r  = sigmoidf_dev(rp);
      float g  = tanhf(gp * r + XRd[rb * HDIM + j]);
      float z  = sigmoidf_dev(zp);
      float ho = (float)lH[b * HDIM + j];
      float hn = z * ho + (1.0f - z) * g;
      _Float16 hv = (_Float16)hn;
      Hping[(size_t)(t & 1) * NB * HDIM + (size_t)b * HDIM + j] = hv;
      Yd[rb * HDIM + j] = hv;
    }

    // Device-wide step barrier (monotonic counter, release/acquire)
    __threadfence();
    __syncthreads();
    if (t < T - 1) {
      if (tid == 0)
        __hip_atomic_fetch_add(ctr, 1u, __ATOMIC_RELEASE, __HIP_MEMORY_SCOPE_AGENT);
      unsigned target = (unsigned)(t + 1) * (unsigned)GRU_NWG;
      while (__hip_atomic_load(ctr, __ATOMIC_ACQUIRE, __HIP_MEMORY_SCOPE_AGENT) < target)
        __builtin_amdgcn_s_sleep(2);
    }
  }
}

// ---------------------------------------------------------------------------
// Final FC: concat(h_f[T-1], h_r[0]) (32x1024) @ Wfc (1024x512) + bfc
// ---------------------------------------------------------------------------
__global__ void gather_fcin(const _Float16* __restrict__ Yf,
                            const _Float16* __restrict__ Yr,
                            _Float16* __restrict__ fcin) {
  int i = blockIdx.x * blockDim.x + threadIdx.x;   // 32*1024
  if (i >= NB * H2DIM) return;
  int b = i >> 10, j = i & 1023;
  fcin[i] = (j < 512) ? Yf[((size_t)(T_STEPS - 1) * NB + b) * HDIM + j]
                      : Yr[(size_t)b * HDIM + (j - 512)];
}

__global__ __launch_bounds__(256, 2) void fc_gemm(
    const _Float16* __restrict__ fcin,   // [32][1024] f16 row-major
    const _Float16* __restrict__ Wt,     // [512][1024] f16 col-major
    const float* __restrict__ bias, float* __restrict__ out) {  // [32][512]
  const int tid = threadIdx.x, lane = tid & 31, wave = tid >> 5;
  const int nBase = blockIdx.x * 64;
  const int m0 = (wave >> 2) * 16;
  const int n0 = nBase + (wave & 3) * 16;
  v8f acc = {};
#pragma unroll 4
  for (int k0 = 0; k0 < H2DIM; k0 += 32) {
    v16h a = frag_a_rm(fcin, H2DIM, m0, k0, lane);
    v16h b = frag_b_cm(Wt, H2DIM, n0, k0, lane);
    acc = wmma16(a, b, acc);
  }
  int n = n0 + (lane & 15);
  float bv = bias[n];
#pragma unroll
  for (int v = 0; v < 8; ++v) {
    int m = m0 + v + 8 * (lane >> 4);
    out[(size_t)m * ODIM + n] = acc[v] + bv;
  }
}

// ---------------------------------------------------------------------------
// Host orchestration
// ---------------------------------------------------------------------------
extern "C" void kernel_launch(void* const* d_in, const int* in_sizes, int n_in,
                              void* d_out, int out_size, void* d_ws,
                              size_t ws_size, hipStream_t stream) {
  (void)in_sizes; (void)n_in; (void)out_size; (void)ws_size;

  const float* x   = (const float*)d_in[0];
  const float* Wxh = (const float*)d_in[1];
  const float* bxh = (const float*)d_in[2];
  const float* Whh = (const float*)d_in[3];
  const float* bhh = (const float*)d_in[4];
  const float* Wxr = (const float*)d_in[5];
  const float* bxr = (const float*)d_in[6];
  const float* Whr = (const float*)d_in[7];
  const float* bhr = (const float*)d_in[8];
  const float* Wfc = (const float*)d_in[9];
  const float* bfc = (const float*)d_in[10];
  float* out = (float*)d_out;

  char* ws = (char*)d_ws;
  size_t off = 0;
  auto take = [&](size_t bytes) {
    char* p = ws + off;
    off = (off + bytes + 255) & ~(size_t)255;
    return p;
  };
  _Float16* seqA = (_Float16*)take((size_t)2 * ROWS * IDIM * 2);   // 8 MB
  _Float16* seqB = (_Float16*)take((size_t)2 * ROWS * IDIM * 2);   // 8 MB
  float*    XHb  = (float*)take((size_t)2 * ROWS * H2DIM * 4);     // 32 MB
  float*    XRb  = (float*)take((size_t)2 * ROWS * HDIM * 4);      // 16 MB
  _Float16* WxhT = (_Float16*)take((size_t)4 * H2DIM * IDIM * 2);  // 4 MB
  _Float16* WhhT = (_Float16*)take((size_t)4 * H2DIM * HDIM * 2);  // 4 MB
  _Float16* WxrT = (_Float16*)take((size_t)4 * HDIM * IDIM * 2);   // 2 MB
  _Float16* WhrT = (_Float16*)take((size_t)4 * HDIM * HDIM * 2);   // 2 MB
  _Float16* WfcT = (_Float16*)take((size_t)ODIM * H2DIM * 2);      // 1 MB
  _Float16* Hbuf = (_Float16*)take((size_t)2 * 2 * NB * HDIM * 2); // 128 KB
  _Float16* fcin = (_Float16*)take((size_t)NB * H2DIM * 2);        // 64 KB
  unsigned* ctr  = (unsigned*)take(256);

  (void)hipFuncSetAttribute(reinterpret_cast<const void*>(gru_step_all),
                            hipFuncAttributeMaxDynamicSharedMemorySize,
                            GRU_SMEM_BYTES);

  init_ctr<<<1, 64, 0, stream>>>(ctr);
  pack_x<<<(T_STEPS * NB * IDIM + 255) / 256, 256, 0, stream>>>(x, seqA,
      seqA + (size_t)ROWS * IDIM);

  for (int l = 0; l < 2; ++l)
    for (int d = 0; d < 2; ++d) {
      size_t idx = (size_t)(l * 2 + d);
      pack_wT<<<(IDIM * H2DIM + 255) / 256, 256, 0, stream>>>(
          Wxh + idx * IDIM * H2DIM, WxhT + idx * H2DIM * IDIM, IDIM, H2DIM);
      pack_wT<<<(HDIM * H2DIM + 255) / 256, 256, 0, stream>>>(
          Whh + idx * HDIM * H2DIM, WhhT + idx * H2DIM * HDIM, HDIM, H2DIM);
      pack_wT<<<(IDIM * HDIM + 255) / 256, 256, 0, stream>>>(
          Wxr + idx * IDIM * HDIM, WxrT + idx * HDIM * IDIM, IDIM, HDIM);
      pack_wT<<<(HDIM * HDIM + 255) / 256, 256, 0, stream>>>(
          Whr + idx * HDIM * HDIM, WhrT + idx * HDIM * HDIM, HDIM, HDIM);
    }
  pack_wT<<<(H2DIM * ODIM + 255) / 256, 256, 0, stream>>>(Wfc, WfcT, H2DIM, ODIM);

  for (int l = 0; l < 2; ++l) {
    const _Float16* inSeq = (l == 0) ? seqA : seqB;
    _Float16* outSeq      = (l == 0) ? seqB : seqA;
    for (int d = 0; d < 2; ++d) {
      size_t idx = (size_t)(l * 2 + d);
      gemm_xproj<<<dim3(H2DIM / 128, ROWS / 64), 256, 0, stream>>>(
          inSeq + (size_t)d * ROWS * IDIM, WxhT + idx * H2DIM * IDIM,
          bxh + idx * H2DIM, XHb + (size_t)d * ROWS * H2DIM, ROWS, H2DIM, IDIM);
      gemm_xproj<<<dim3(HDIM / 128, ROWS / 64), 256, 0, stream>>>(
          inSeq + (size_t)d * ROWS * IDIM, WxrT + idx * HDIM * IDIM,
          bxr + idx * HDIM, XRb + (size_t)d * ROWS * HDIM, ROWS, HDIM, IDIM);
    }
    gru_step_all<<<GRU_NWG, GRU_THREADS, GRU_SMEM_BYTES, stream>>>(
        WhhT + (size_t)l * 2 * H2DIM * HDIM, WhrT + (size_t)l * 2 * HDIM * HDIM,
        bhh + (size_t)l * 2 * H2DIM, bhr + (size_t)l * 2 * HDIM,
        XHb, XRb, outSeq, Hbuf, ctr + l, T_STEPS);
  }

  // Layer-1 outputs live in seqA (dir0 fwd, dir1 rev)
  gather_fcin<<<(NB * H2DIM + 255) / 256, 256, 0, stream>>>(
      seqA, seqA + (size_t)ROWS * HDIM, fcin);
  fc_gemm<<<ODIM / 64, 256, 0, stream>>>(fcin, WfcT, bfc, out);
}
